// GraphFeaturesStackPad_80101140070614
// MI455X (gfx1250) — matrix operand order repaired and checked by reading, not compiled, not run on hardware
//
#include <hip/hip_runtime.h>

// GraphFeaturesStackPad for MI455X (gfx1250, wave32, WMMA f16->f32)
//
// out[g] = (sum_{nodes n of graph g, excluding last} sigmoid(x_n@Wg + bg) * (x_n@Wu + bu)) @ Wf + bf

typedef __attribute__((ext_vector_type(16))) _Float16 v16h;
typedef __attribute__((ext_vector_type(8)))  _Float16 v8h;
typedef __attribute__((ext_vector_type(4)))  _Float16 v4h;
typedef __attribute__((ext_vector_type(8)))  float    v8f;
typedef __attribute__((ext_vector_type(4)))  float    v4f;

#define LDS_WSTRIDE 136   // halves per transposed-weight row: 272B = 17*16 (keeps b128 align, spreads banks)

#define WMMA_F16(A, B, C) \
    __builtin_amdgcn_wmma_f32_16x16x32_f16(false, (A), false, (B), (short)0, (C), false, false)

// A fragment, contiguous k order used here (same as ldsb): lane L<16: row M=L, K = kb..kb+15 (kb=0);
// lane L>=16: row M=L-16, kb=16.
__device__ __forceinline__ v16h make_afrag(const float* __restrict__ rowptr, int c, int lane, bool valid) {
    v16h a;
    if (valid) {
        const int kb = c * 32 + ((lane & 16) ? 16 : 0);
        const v4f* p = (const v4f*)(rowptr + kb);
        alignas(16) float tmp[16];
        *(v4f*)&tmp[0]  = p[0];   // K = kb .. kb+3
        *(v4f*)&tmp[4]  = p[1];   // K = kb+4 .. kb+7
        *(v4f*)&tmp[8]  = p[2];   // K = kb+8 .. kb+11
        *(v4f*)&tmp[12] = p[3];   // K = kb+12 .. kb+15
        #pragma unroll
        for (int i = 0; i < 16; ++i) a[i] = (_Float16)tmp[i];
    } else {
        #pragma unroll
        for (int i = 0; i < 16; ++i) a[i] = (_Float16)0.0f;
    }
    return a;
}

// B fragment from LDS (weights stored transposed: wt[n*LDS_WSTRIDE + k], f16).
// contiguous k order used here (same as make_afrag): lanes 0-15: N=16j+lane, K=32c..32c+15; lanes 16-31: N=16j+lane-16, K=32c+16..32c+31.
__device__ __forceinline__ v16h ldsb(const _Float16* __restrict__ wt, int j, int c, int lane) {
    const _Float16* p = wt + (j * 16 + (lane & 15)) * LDS_WSTRIDE + c * 32 + ((lane & 16) ? 16 : 0);
    v8h lo = *(const v8h*)p;
    v8h hi = *(const v8h*)(p + 8);
    return __builtin_shufflevector(lo, hi, 0,1,2,3,4,5,6,7,8,9,10,11,12,13,14,15);
}

__global__ __launch_bounds__(256) void gfs_kernel(
    const float* __restrict__ nf,
    const float* __restrict__ Wup,   const float* __restrict__ bup,
    const float* __restrict__ Wgate, const float* __restrict__ bgate,
    const float* __restrict__ Wfunc, const float* __restrict__ bfunc,
    const int*   __restrict__ starts,
    float*       __restrict__ out,
    int G)
{
    extern __shared__ char smem[];
    _Float16* wtu = (_Float16*)smem;                 // 136*128 halves, transposed W_up
    _Float16* wtg = wtu + LDS_WSTRIDE * 128;         // 136*128 halves, transposed W_gate
    _Float16* wf  = wtg + LDS_WSTRIDE * 128;         // 128*128 halves, W_func row-major
    float* bu   = (float*)(wf + 128 * 128);
    float* bg   = bu + 128;
    float* bf   = bg + 128;
    float* gsum = bf + 128;                          // 8 waves x 128 f32, wave-private column sums

    const int tid = threadIdx.x;

    // ---- Preload weights into LDS (once per workgroup) ----
    for (int idx = tid; idx < 128 * 128; idx += blockDim.x) {
        const int k = idx >> 7, n = idx & 127;       // idx = k*128 + n (row-major [k][n])
        wtu[n * LDS_WSTRIDE + k] = (_Float16)Wup[idx];
        wtg[n * LDS_WSTRIDE + k] = (_Float16)Wgate[idx];
        wf[idx]                  = (_Float16)Wfunc[idx];
    }
    if (tid < 128) { bu[tid] = bup[tid]; bg[tid] = bgate[tid]; bf[tid] = bfunc[tid]; }
    __syncthreads();

    const int lane       = tid & 31;
    const int waveId     = blockIdx.x * (blockDim.x >> 5) + (tid >> 5);
    const int waveStride = gridDim.x * (blockDim.x >> 5);
    float* gsumw = gsum + (tid >> 5) * 128;          // this wave's private accumulator

    // ---- One graph per wave, grid-strided ----
    for (int g = waveId; g < G; g += waveStride) {
        const int s0    = starts[g];
        const int keepN = starts[g + 1] - s0 - 1;    // exclude last node of each graph (>=1)

        // zero wave-private accumulator (wave-private: no barrier needed, ds ops in-order per wave)
        #pragma unroll
        for (int i = 0; i < 4; ++i) gsumw[lane + 32 * i] = 0.0f;

        for (int t = 0; t * 16 < keepN; ++t) {
            const int  lrow   = t * 16 + (lane & 15);
            const bool rvalid = lrow < keepN;
            const float* rowptr = nf + (size_t)(s0 + lrow) * 128;

            v16h a[4];
            #pragma unroll
            for (int c = 0; c < 4; ++c) a[c] = make_afrag(rowptr, c, lane, rvalid);

            const int rowbase = t * 16 + ((lane & 16) ? 8 : 0);  // C layout: VGPR e -> M = e (+8 hi-half)

            #pragma unroll 1   // keep rolled: bounds VGPR pressure, keeps B-frags in LDS (no scratch spill)
            for (int j = 0; j < 8; ++j) {
                v8f up = {}; v8f gt = {};
                #pragma unroll
                for (int c = 0; c < 4; ++c) {
                    up = WMMA_F16(a[c], ldsb(wtu, j, c, lane), up);
                    gt = WMMA_F16(a[c], ldsb(wtg, j, c, lane), gt);
                }
                const int   ncol = j * 16 + (lane & 15);
                const float bub = bu[ncol], bgb = bg[ncol];
                float part = 0.0f;
                #pragma unroll
                for (int e = 0; e < 8; ++e) {
                    // fast sigmoid: v_exp_f32 + v_rcp_f32 (TRANS ops co-issue with WMMA)
                    const float s   = __builtin_amdgcn_rcpf(1.0f + __expf(-(gt[e] + bgb)));
                    const float val = s * (up[e] + bub);
                    part += (rowbase + e < keepN) ? val : 0.0f;   // branchless keep-mask
                }
                part += __shfl_xor(part, 16, 32);     // rows 0-7 (lanes 0-15) + rows 8-15 (lanes 16-31)
                if (lane < 16) gsumw[ncol] += part;   // wave-private, one writer per column
            }
        }

        // ---- Stage 2: out[g] = gsum @ W_func + b_func (per-wave GEMV, f16 weights from LDS) ----
        float o0 = bf[lane * 4 + 0];
        float o1 = bf[lane * 4 + 1];
        float o2 = bf[lane * 4 + 2];
        float o3 = bf[lane * 4 + 3];
        const v4h* wf4 = (const v4h*)wf;              // 4 halves per lane per k-row
        #pragma unroll 8
        for (int k = 0; k < 128; ++k) {
            const float w = gsumw[k];                 // LDS broadcast read
            const v4h  h = wf4[k * 32 + lane];
            o0 += w * (float)h[0];
            o1 += w * (float)h[1];
            o2 += w * (float)h[2];
            o3 += w * (float)h[3];
        }
        v4f o = { o0, o1, o2, o3 };
        *(v4f*)(out + (size_t)g * 128 + lane * 4) = o;
    }
}

extern "C" void kernel_launch(void* const* d_in, const int* in_sizes, int n_in,
                              void* d_out, int out_size, void* d_ws, size_t ws_size,
                              hipStream_t stream) {
    (void)n_in; (void)out_size; (void)d_ws; (void)ws_size;
    const float* nf     = (const float*)d_in[0];
    const float* Wup    = (const float*)d_in[1];
    const float* bup    = (const float*)d_in[2];
    const float* Wgate  = (const float*)d_in[3];
    const float* bgate  = (const float*)d_in[4];
    const float* Wfunc  = (const float*)d_in[5];
    const float* bfunc  = (const float*)d_in[6];
    const int*   starts = (const int*)d_in[7];
    float* out = (float*)d_out;

    const int G = in_sizes[7] - 1;

    const int threads = 256;                 // 8 waves/WG on wave32
    int blocks = 512;                        // ~4096 waves -> ~6 graphs/wave, amortizes LDS weight preload
    const int wavesPerBlock = threads / 32;
    const int maxBlocks = (G + wavesPerBlock - 1) / wavesPerBlock;
    if (blocks > maxBlocks) blocks = maxBlocks;
    if (blocks < 1) blocks = 1;

    // LDS: two transposed f16 weight mats (stride 136) + f16 W_func + 3 f32 bias vectors + 8x128 f32 accums
    const size_t shmem = (size_t)(2 * LDS_WSTRIDE * 128 + 128 * 128) * 2 + 3 * 128 * 4 + 8 * 128 * 4;

    gfs_kernel<<<blocks, threads, shmem, stream>>>(nf, Wup, bup, Wgate, bgate,
                                                   Wfunc, bfunc, starts, out, G);
}
